// CosBlock_5188320494306
// MI455X (gfx1250) — matrix operand order repaired and checked by reading, not compiled, not run on hardware
//
#include <hip/hip_runtime.h>
#include <hip/hip_bf16.h>

// ---------------------------------------------------------------------------
// CosBlock on gfx1250: LN1 -> cos linear attention -> +x -> LN2 -> MLP -> +x
// GEMMs: v_wmma_f32_16x16x32_bf16, double-buffered LDS fed by BOTH CDNA5
// async movers: A tiles via the Tensor Data Mover (tensor_load_to_lds,
// TENSORcnt) and B tiles via global_load_async_to_lds_b128 (ASYNCcnt).
// Weights packed pre-transposed (N x K) so tiles are contiguous 2D blocks.
// ---------------------------------------------------------------------------

typedef __attribute__((ext_vector_type(16))) __bf16 v16bf;
typedef __attribute__((ext_vector_type(8)))  float  v8f;
typedef __attribute__((ext_vector_type(8)))  unsigned int v8u;
typedef __attribute__((ext_vector_type(4)))  unsigned int v4u;
typedef __attribute__((ext_vector_type(8)))  int v8i;
typedef __attribute__((ext_vector_type(4)))  int v4i;

#define D_HIDDEN 1024
#define N_HEADS  16
#define D_HEAD   64
#define D_MLP    4096
#define BATCH    4
#define SEQ      2048
#define BT       (BATCH*SEQ)      // 8192 rows
#define NCHUNK   32
#define CHUNK    (SEQ/NCHUNK)     // 64

__device__ __forceinline__ unsigned short f2bf(float f) {
  unsigned u = __builtin_bit_cast(unsigned, f);
  u = (u + 0x7FFFu + ((u >> 16) & 1u)) >> 16;
  return (unsigned short)u;
}
__device__ __forceinline__ float bf2f(unsigned short h) {
  unsigned u = ((unsigned)h) << 16;
  return __builtin_bit_cast(float, u);
}
__device__ __forceinline__ float gelu_tanh(float x) {
  float x3 = x * x * x;
  float t = tanhf(0.7978845608028654f * (x + 0.044715f * x3));
  return 0.5f * x * (1.0f + t);
}

// Issue two async 16B global->LDS copies covering one 32B row-chunk.
// ISA: instruction offset is added to BOTH the LDS and global addresses.
__device__ __forceinline__ void stage32(unsigned lds_off, const unsigned short* g) {
  unsigned long long ga = (unsigned long long)(size_t)g;
  asm volatile("global_load_async_to_lds_b128 %0, %1, off"
               :: "v"(lds_off), "v"(ga) : "memory");
  asm volatile("global_load_async_to_lds_b128 %0, %1, off offset:16"
               :: "v"(lds_off), "v"(ga) : "memory");
}

// ---------------------------------------------------------------------------
// Tensor Data Mover descriptor (D#) for a 128-row x 32-elem bf16 tile of a
// row-major (rows x K) matrix. Group1 is K-dependent only; group0 carries the
// per-tile LDS and global addresses. 2D walk => groups 2/3 zero.
// ---------------------------------------------------------------------------
__device__ __forceinline__ v8i tdm_g1(int K) {
  v8i g;
  g[0] = (int)(1u << 16);                               // wg_mask=0, data_size=2B
  g[1] = (int)((((unsigned)K) & 0xFFFFu) << 16);        // tensor_dim0[15:0] @63:48
  g[2] = (int)((((unsigned)K) >> 16) & 0xFFFFu);        // tensor_dim0[31:16]; dim1 lo16=0
  g[3] = (int)(0x4000u | (32u << 16));                  // dim1 hi16 (2^30 rows); tile_dim0=32
  g[4] = 128;                                           // tile_dim1=128, tile_dim2=0
  g[5] = K;                                             // tensor_dim0_stride[31:0]
  g[6] = 0;                                             // stride hi, dim1_stride lo
  g[7] = 0;
  return g;
}
__device__ __forceinline__ void tdm_load(unsigned lds_addr, unsigned long long ga, v8i g1) {
  v4u g0;
  g0[0] = 1u;                                           // count=1 (valid, user mode)
  g0[1] = lds_addr;                                     // lds_addr @63:32
  g0[2] = (unsigned)ga;                                 // global_addr[31:0]
  g0[3] = (unsigned)((ga >> 32) & 0x01FFFFFFull) | 0x80000000u; // addr[56:32]; type=2
  v4i z4 = {0, 0, 0, 0};
#if defined(__clang_major__) && (__clang_major__ >= 23)
  v8i z8 = {0, 0, 0, 0, 0, 0, 0, 0};
  __builtin_amdgcn_tensor_load_to_lds(g0, g1, z4, z4, z8, 0);
#else
  __builtin_amdgcn_tensor_load_to_lds(g0, g1, z4, z4, 0);
#endif
}

// ---------------------------------------------------------------------------
// Weight packing (fp32 -> bf16). All GEMM B operands are packed TRANSPOSED
// (N x K row-major) so kernel-side staging is contiguous.
// ---------------------------------------------------------------------------
__global__ void pack_wv(const float* __restrict__ W, unsigned short* __restrict__ out) {
  int idx = blockIdx.x * blockDim.x + threadIdx.x;          // i*65536 + d*64 + h
  if (idx >= N_HEADS * D_HIDDEN * D_HEAD) return;
  int i = idx >> 16;
  int rem = idx & 0xFFFF;
  int d = rem >> 6;
  int h = rem & 63;
  // WvBT[n = i*64+h][k = d]
  out[(size_t)(i * 64 + h) * D_HIDDEN + d] = f2bf(W[idx]);
}
__global__ void pack_wo(const float* __restrict__ W, unsigned short* __restrict__ out) {
  int idx = blockIdx.x * blockDim.x + threadIdx.x;          // d*1024 + h*16 + i
  if (idx >= D_HIDDEN * D_HEAD * N_HEADS) return;
  int d = idx >> 10;
  int rem = idx & 1023;
  int h = rem >> 4;
  int i = rem & 15;
  // WoBT[n = d][k = i*64+h]
  out[(size_t)d * D_HIDDEN + i * 64 + h] = f2bf(W[idx]);
}
// out[c*R + r] = in[r*C + c]  (B^T pack for plain row-major weights)
__global__ void pack_transpose(const float* __restrict__ W, unsigned short* __restrict__ out,
                               int R, int C) {
  int idx = blockIdx.x * blockDim.x + threadIdx.x;
  if (idx >= R * C) return;
  int r = idx / C, c = idx - r * C;
  out[(size_t)c * R + r] = f2bf(W[idx]);
}

// ---------------------------------------------------------------------------
// LayerNorm -> bf16 row output. One 256-thread block per row (D=1024).
// ---------------------------------------------------------------------------
__global__ __launch_bounds__(256) void ln_bf16(const float* __restrict__ x,
                                               const float* __restrict__ w,
                                               const float* __restrict__ b,
                                               unsigned short* __restrict__ y) {
  __shared__ float red[256];
  const int row = blockIdx.x, tid = threadIdx.x;
  const float* xr = x + (size_t)row * D_HIDDEN;
  float xs[4];
  float s = 0.f;
#pragma unroll
  for (int j = 0; j < 4; ++j) { xs[j] = xr[tid + 256 * j]; s += xs[j]; }
  red[tid] = s; __syncthreads();
  for (int o = 128; o > 0; o >>= 1) { if (tid < o) red[tid] += red[tid + o]; __syncthreads(); }
  float mu = red[0] * (1.0f / D_HIDDEN);
  __syncthreads();
  float s2 = 0.f;
#pragma unroll
  for (int j = 0; j < 4; ++j) { float d = xs[j] - mu; s2 += d * d; }
  red[tid] = s2; __syncthreads();
  for (int o = 128; o > 0; o >>= 1) { if (tid < o) red[tid] += red[tid + o]; __syncthreads(); }
  float inv = rsqrtf(red[0] * (1.0f / D_HIDDEN) + 1e-5f);
#pragma unroll
  for (int j = 0; j < 4; ++j) {
    int c = tid + 256 * j;
    y[(size_t)row * D_HIDDEN + c] = f2bf((xs[j] - mu) * inv * w[c] + b[c]);
  }
}

// ---------------------------------------------------------------------------
// q/k scalar projections per head + tanh*pi/4 + sin/cos. One block per row.
// ---------------------------------------------------------------------------
__global__ __launch_bounds__(256) void qk_kernel(const unsigned short* __restrict__ y,
                                                 const float* __restrict__ WQ,
                                                 const float* __restrict__ WK,
                                                 float* __restrict__ cq, float* __restrict__ sq,
                                                 float* __restrict__ ck, float* __restrict__ sk) {
  __shared__ float sy[D_HIDDEN];
  __shared__ float pr[512];
  const int row = blockIdx.x, tid = threadIdx.x;
#pragma unroll
  for (int j = 0; j < 4; ++j) {
    int c = tid + 256 * j;
    sy[c] = bf2f(y[(size_t)row * D_HIDDEN + c]);
  }
  __syncthreads();
  const int head = tid >> 4, seg = tid & 15;
  const float* wq = WQ + head * D_HIDDEN;
  const float* wk = WK + head * D_HIDDEN;
  float aq = 0.f, ak = 0.f;
  const int d0 = seg * 64;
#pragma unroll 8
  for (int d = d0; d < d0 + 64; ++d) { aq += sy[d] * wq[d]; ak += sy[d] * wk[d]; }
  pr[tid] = aq; pr[256 + tid] = ak;
  __syncthreads();
  if (seg == 0) {
    float tq = 0.f, tk = 0.f;
#pragma unroll
    for (int j = 0; j < 16; ++j) { tq += pr[head * 16 + j]; tk += pr[256 + head * 16 + j]; }
    float qv = tanhf(tq) * 0.7853981633974483f;
    float kv = tanhf(tk) * 0.7853981633974483f;
    float cs, sn;
    int o = row * N_HEADS + head;
    sincosf(qv, &sn, &cs); cq[o] = cs; sq[o] = sn;
    sincosf(kv, &sn, &cs); ck[o] = cs; sk[o] = sn;
  }
}

// ---------------------------------------------------------------------------
// BF16 WMMA GEMM: C(MxN) = A(MxK) * Bt^T  with Bt given as (N x K) row-major.
// Block tile 128x128, 8 waves of 32x64, K-tile 32. Double-buffered LDS:
// A tile staged by the Tensor Data Mover (wave 0, TENSORcnt), B tile staged
// by per-lane global_load_async_to_lds_b128 (ASYNCcnt). MODE epilogues:
//  1: fp32 acc + resid         2: bf16 gelu(acc + bias)
//  3: fp32 acc + bias + resid  4: bf16 acc
// ---------------------------------------------------------------------------
template <int MODE>
__global__ __launch_bounds__(256) void gemm_bf16_wmma(
    const unsigned short* __restrict__ A, const unsigned short* __restrict__ Bt,
    const float* __restrict__ bias, const float* __restrict__ resid,
    void* __restrict__ outv, int M, int N, int K) {
  __shared__ unsigned short lA[2][128 * 32];       // [buf][m][k]
  __shared__ unsigned short lB[2][128 * 32];       // [buf][n][k]
  const int tid = threadIdx.x;
  const int lane = tid & 31, w = tid >> 5;
  const int wm = w & 3, wn = w >> 2;               // wave origin: (wm*32, wn*64)
  const int m0 = blockIdx.y * 128, n0 = blockIdx.x * 128;

  v8f acc[2][4];
#pragma unroll
  for (int i = 0; i < 2; ++i)
#pragma unroll
    for (int j = 0; j < 4; ++j)
#pragma unroll
      for (int r = 0; r < 8; ++r) acc[i][j][r] = 0.0f;

  // --- A tiles via TDM (wave 0 issues; descriptor fields are wave-uniform)
  const unsigned long long Abase =
      (unsigned long long)(size_t)(A + (size_t)m0 * K);
  const unsigned ldsA[2] = { (unsigned)(size_t)&lA[0][0],
                             (unsigned)(size_t)&lA[1][0] };
  const v8i g1 = tdm_g1(K);

  // --- B tiles via per-lane async copies: one 32B half-row per thread
  const int sr = tid >> 1, sh = (tid & 1) * 16;    // row 0..127, half offset
  const unsigned short* Bg = Bt + (size_t)(n0 + sr) * K + sh;
  const unsigned stB[2] = {
      (unsigned)(size_t)&lB[0][sr * 32 + sh], (unsigned)(size_t)&lB[1][sr * 32 + sh] };

  const int mi = lane & 15, half = lane >> 4;

  // prologue: fill buffer 0
  if (w == 0) tdm_load(ldsA[0], Abase, g1);
  stage32(stB[0], Bg);

  int cur = 0;
  for (int k0 = 0; k0 < K; k0 += 32, cur ^= 1) {
    if (w == 0) __builtin_amdgcn_s_wait_tensorcnt(0);     // A tile landed
    asm volatile("s_wait_asynccnt 0x0" ::: "memory");     // B tile landed
    __syncthreads();                                      // buffer `cur` ready
    if (k0 + 32 < K) {                                    // stage next tile
      if (w == 0) tdm_load(ldsA[cur ^ 1], Abase + (size_t)(k0 + 32) * 2, g1);
      stage32(stB[cur ^ 1], Bg + k0 + 32);
    }

    // A fragments: ISA 16-bit A 16x32 lane layout
    v8u afr[2];
#pragma unroll
    for (int mt = 0; mt < 2; ++mt) {
      const int mrow = wm * 32 + mt * 16 + mi;
#pragma unroll
      for (int v = 0; v < 8; ++v) {
        const int kb = ((v >= 4) ? 16 : 0) + (v & 3) * 2 + half * 8;
        afr[mt][v] = *(const unsigned int*)&lA[cur][mrow * 32 + kb];
      }
    }
    // B fragments: ISA 16-bit B 32x16 lane layout (K pairs contiguous in [n][k])
    v8u bfr[4];
#pragma unroll
    for (int nt = 0; nt < 4; ++nt) {
      const int ncol = wn * 64 + nt * 16 + mi;
#pragma unroll
      for (int v = 0; v < 8; ++v) {
        const int kb = ((v >= 4) ? 16 : 0) + (v & 3) * 2 + half * 8;
        bfr[nt][v] = *(const unsigned int*)&lB[cur][ncol * 32 + kb];
      }
    }
#pragma unroll
    for (int mt = 0; mt < 2; ++mt)
#pragma unroll
      for (int nt = 0; nt < 4; ++nt)
        acc[mt][nt] = __builtin_amdgcn_wmma_f32_16x16x32_bf16(
            false, __builtin_bit_cast(v16bf, afr[mt]),
            false, __builtin_bit_cast(v16bf, bfr[nt]),
            (short)0, acc[mt][nt], false, false);

    __syncthreads();   // all reads of `cur` done before it is restaged next iter
  }

  float* outf = (float*)outv;
  unsigned short* outh = (unsigned short*)outv;
#pragma unroll
  for (int mt = 0; mt < 2; ++mt)
#pragma unroll
    for (int nt = 0; nt < 4; ++nt)
#pragma unroll
      for (int r = 0; r < 8; ++r) {
        const size_t m = (size_t)m0 + wm * 32 + mt * 16 + half * 8 + r;
        const size_t n = (size_t)n0 + wn * 64 + nt * 16 + mi;
        float v = acc[mt][nt][r];
        if (MODE == 1) v += resid[m * N + n];
        if (MODE == 2) v = gelu_tanh(v + bias[n]);
        if (MODE == 3) v += bias[n] + resid[m * N + n];
        if (MODE == 2 || MODE == 4) outh[m * N + n] = f2bf(v);
        else                        outf[m * N + n] = v;
      }
}

// ---------------------------------------------------------------------------
// 3-pass chunked causal scan over T (inclusive cumsum of cos_k*V, sin_k*V,
// cos_k, sin_k), then heads = (cq*cKV + sq*sKV)/(cq*cK + sq*sK + eps).
// ---------------------------------------------------------------------------
__global__ __launch_bounds__(64) void scan_partial(
    const unsigned short* __restrict__ V, const float* __restrict__ ck,
    const float* __restrict__ sk, float* __restrict__ ckv_sum,
    float* __restrict__ skv_sum, float* __restrict__ cks_sum,
    float* __restrict__ sks_sum) {
  const int bi = blockIdx.x >> 5, c = blockIdx.x & 31;
  const int b = bi >> 4, i = bi & 15, d = threadIdx.x;
  float ckv = 0.f, skv = 0.f, cs = 0.f, ss = 0.f;
  const int t0 = c * CHUNK;
  for (int t = t0; t < t0 + CHUNK; ++t) {
    const int row = b * SEQ + t;
    const float ckt = ck[row * N_HEADS + i], skt = sk[row * N_HEADS + i];
    const float v = bf2f(V[(size_t)row * D_HIDDEN + i * D_HEAD + d]);
    ckv += ckt * v; skv += skt * v; cs += ckt; ss += skt;
  }
  const int o = (bi * NCHUNK + c) * D_HEAD + d;
  ckv_sum[o] = ckv; skv_sum[o] = skv;
  if (d == 0) { cks_sum[bi * NCHUNK + c] = cs; sks_sum[bi * NCHUNK + c] = ss; }
}

__global__ __launch_bounds__(64) void scan_offsets(
    const float* __restrict__ ckv_sum, const float* __restrict__ skv_sum,
    const float* __restrict__ cks_sum, const float* __restrict__ sks_sum,
    float* __restrict__ ckv_off, float* __restrict__ skv_off,
    float* __restrict__ cks_off, float* __restrict__ sks_off) {
  const int bi = blockIdx.x, d = threadIdx.x;
  float aC = 0.f, aS = 0.f, bC = 0.f, bS = 0.f;
  for (int c = 0; c < NCHUNK; ++c) {
    const int o = (bi * NCHUNK + c) * D_HEAD + d;
    ckv_off[o] = aC; skv_off[o] = aS;
    if (d == 0) { cks_off[bi * NCHUNK + c] = bC; sks_off[bi * NCHUNK + c] = bS; }
    aC += ckv_sum[o]; aS += skv_sum[o];
    bC += cks_sum[bi * NCHUNK + c]; bS += sks_sum[bi * NCHUNK + c];
  }
}

__global__ __launch_bounds__(64) void scan_apply(
    const unsigned short* __restrict__ V, const float* __restrict__ ck,
    const float* __restrict__ sk, const float* __restrict__ cq,
    const float* __restrict__ sq, const float* __restrict__ ckv_off,
    const float* __restrict__ skv_off, const float* __restrict__ cks_off,
    const float* __restrict__ sks_off, unsigned short* __restrict__ heads) {
  const int bi = blockIdx.x >> 5, c = blockIdx.x & 31;
  const int b = bi >> 4, i = bi & 15, d = threadIdx.x;
  float ckv = ckv_off[(bi * NCHUNK + c) * D_HEAD + d];
  float skv = skv_off[(bi * NCHUNK + c) * D_HEAD + d];
  float cK = cks_off[bi * NCHUNK + c];
  float sK = sks_off[bi * NCHUNK + c];
  const int t0 = c * CHUNK;
  for (int t = t0; t < t0 + CHUNK; ++t) {
    const int row = b * SEQ + t;
    const float ckt = ck[row * N_HEADS + i], skt = sk[row * N_HEADS + i];
    const float v = bf2f(V[(size_t)row * D_HIDDEN + i * D_HEAD + d]);
    ckv += ckt * v; skv += skt * v; cK += ckt; sK += skt;
    const float cqv = cq[row * N_HEADS + i], sqv = sq[row * N_HEADS + i];
    const float h = (cqv * ckv + sqv * skv) / (cqv * cK + sqv * sK + 1e-6f);
    heads[(size_t)row * D_HIDDEN + i * D_HEAD + d] = f2bf(h);
  }
}

// ---------------------------------------------------------------------------
extern "C" void kernel_launch(void* const* d_in, const int* in_sizes, int n_in,
                              void* d_out, int out_size, void* d_ws, size_t ws_size,
                              hipStream_t stream) {
  const float* x     = (const float*)d_in[0];
  const float* W_Q   = (const float*)d_in[1];
  const float* W_K   = (const float*)d_in[2];
  const float* W_V   = (const float*)d_in[3];
  const float* W_O   = (const float*)d_in[4];
  const float* ln1_w = (const float*)d_in[5];
  const float* ln1_b = (const float*)d_in[6];
  const float* ln2_w = (const float*)d_in[7];
  const float* ln2_b = (const float*)d_in[8];
  const float* W1    = (const float*)d_in[9];
  const float* b1    = (const float*)d_in[10];
  const float* W2    = (const float*)d_in[11];
  const float* b2    = (const float*)d_in[12];
  float* out = (float*)d_out;

  char* ws = (char*)d_ws;
  size_t off = 0;
  auto carve = [&](size_t bytes) -> char* {
    char* p = ws + off;
    off = (off + bytes + 255) & ~(size_t)255;
    return p;
  };
  unsigned short* y1    = (unsigned short*)carve((size_t)BT * D_HIDDEN * 2);
  unsigned short* Vb    = (unsigned short*)carve((size_t)BT * D_HIDDEN * 2);
  unsigned short* heads = (unsigned short*)carve((size_t)BT * D_HIDDEN * 2);
  unsigned short* y2    = (unsigned short*)carve((size_t)BT * D_HIDDEN * 2);
  unsigned short* h1    = (unsigned short*)carve((size_t)BT * D_MLP * 2);
  unsigned short* WvBT  = (unsigned short*)carve((size_t)D_HIDDEN * D_HIDDEN * 2);
  unsigned short* WoBT  = (unsigned short*)carve((size_t)D_HIDDEN * D_HIDDEN * 2);
  unsigned short* W1t   = (unsigned short*)carve((size_t)D_HIDDEN * D_MLP * 2);
  unsigned short* W2t   = (unsigned short*)carve((size_t)D_MLP * D_HIDDEN * 2);
  float* cq = (float*)carve((size_t)BT * N_HEADS * 4);
  float* sq = (float*)carve((size_t)BT * N_HEADS * 4);
  float* ck = (float*)carve((size_t)BT * N_HEADS * 4);
  float* sk = (float*)carve((size_t)BT * N_HEADS * 4);
  float* ckv_sum = (float*)carve((size_t)64 * NCHUNK * D_HEAD * 4);
  float* skv_sum = (float*)carve((size_t)64 * NCHUNK * D_HEAD * 4);
  float* ckv_off = (float*)carve((size_t)64 * NCHUNK * D_HEAD * 4);
  float* skv_off = (float*)carve((size_t)64 * NCHUNK * D_HEAD * 4);
  float* cks_sum = (float*)carve((size_t)64 * NCHUNK * 4);
  float* sks_sum = (float*)carve((size_t)64 * NCHUNK * 4);
  float* cks_off = (float*)carve((size_t)64 * NCHUNK * 4);
  float* sks_off = (float*)carve((size_t)64 * NCHUNK * 4);

  // 1) weight packing (pre-transposed for the GEMM B operand)
  pack_wv<<<(N_HEADS * D_HIDDEN * D_HEAD + 255) / 256, 256, 0, stream>>>(W_V, WvBT);
  pack_wo<<<(D_HIDDEN * D_HEAD * N_HEADS + 255) / 256, 256, 0, stream>>>(W_O, WoBT);
  pack_transpose<<<(D_HIDDEN * D_MLP + 255) / 256, 256, 0, stream>>>(W1, W1t, D_HIDDEN, D_MLP);
  pack_transpose<<<(D_MLP * D_HIDDEN + 255) / 256, 256, 0, stream>>>(W2, W2t, D_MLP, D_HIDDEN);

  // 2) LN1
  ln_bf16<<<BT, 256, 0, stream>>>(x, ln1_w, ln1_b, y1);

  // 3) q/k per-head scalars -> cos/sin
  qk_kernel<<<BT, 256, 0, stream>>>(y1, W_Q, W_K, cq, sq, ck, sk);

  // 4) V projection: Vb = y1 * Wv  (bf16 out)
  gemm_bf16_wmma<4><<<dim3(D_HIDDEN / 128, BT / 128), 256, 0, stream>>>(
      y1, WvBT, nullptr, nullptr, Vb, BT, D_HIDDEN, D_HIDDEN);

  // 5) chunked causal scan -> heads (bf16)
  scan_partial<<<64 * NCHUNK, 64, 0, stream>>>(Vb, ck, sk, ckv_sum, skv_sum, cks_sum, sks_sum);
  scan_offsets<<<64, 64, 0, stream>>>(ckv_sum, skv_sum, cks_sum, sks_sum,
                                      ckv_off, skv_off, cks_off, sks_off);
  scan_apply<<<64 * NCHUNK, 64, 0, stream>>>(Vb, ck, sk, cq, sq,
                                             ckv_off, skv_off, cks_off, sks_off, heads);

  // 6) O projection + residual: x1 = heads * Wo + x   (fp32, lives in d_out)
  gemm_bf16_wmma<1><<<dim3(D_HIDDEN / 128, BT / 128), 256, 0, stream>>>(
      heads, WoBT, nullptr, x, out, BT, D_HIDDEN, D_HIDDEN);

  // 7) LN2 on x1
  ln_bf16<<<BT, 256, 0, stream>>>(out, ln2_w, ln2_b, y2);

  // 8) MLP up + GELU: h1 = gelu(y2 * W1 + b1)  (bf16)
  gemm_bf16_wmma<2><<<dim3(D_MLP / 128, BT / 128), 256, 0, stream>>>(
      y2, W1t, b1, nullptr, h1, BT, D_MLP, D_HIDDEN);

  // 9) MLP down + bias + residual: out = h1 * W2 + b2 + x1  (fp32, in-place on d_out)
  gemm_bf16_wmma<3><<<dim3(D_HIDDEN / 128, BT / 128), 256, 0, stream>>>(
      h1, W2t, b2, out, out, BT, D_HIDDEN, D_MLP);
}
